// LinearDST_71923522339425
// MI455X (gfx1250) — matrix operand order repaired and checked by reading, not compiled, not run on hardware
//
#include <hip/hip_runtime.h>

typedef __bf16 bf16;
typedef __attribute__((ext_vector_type(8)))  __bf16 v8bf;
typedef __attribute__((ext_vector_type(16))) __bf16 v16bf;
typedef __attribute__((ext_vector_type(8)))  float  v8f;
typedef __attribute__((ext_vector_type(4)))  float  v4f;

#ifndef M_PI_F
#define M_PI_F 3.14159265358979323846f
#endif

// ---------------------------------------------------------------------------
// Prepass 1: materialize the DST weight matrix in bf16, row-major [OUT][IN].
// W[o][t] = 2 * norm[o] * sin( (pi/2) * (fc[o]+1)/IN * (2t+1) )
// norm[o] = rsqrt(2*IN*(1 + (o==0)))
// ---------------------------------------------------------------------------
__global__ __launch_bounds__(256) void build_w_bf16(
    const float* __restrict__ fc, bf16* __restrict__ Wb, int OUT, int IN) {
  int idx = blockIdx.x * blockDim.x + threadIdx.x;
  int rowChunks = IN >> 3;
  int total = OUT * rowChunks;
  if (idx >= total) return;
  int o  = idx / rowChunks;
  int t0 = (idx - o * rowChunks) << 3;

  float f    = fc[o];
  float eyep = (o == 0) ? 2.0f : 1.0f;                 // eye_col + 1
  float norm = rsqrtf(2.0f * (float)IN * eyep);
  float c    = 0.5f * M_PI_F * (f + 1.0f) / (float)IN; // phase slope
  float s    = 2.0f * norm;

  v8bf w;
#pragma unroll
  for (int j = 0; j < 8; ++j) {
    float t = (float)(t0 + j);
    // args can be O(1e4): use full-precision sinf (proper range reduction)
    w[j] = (__bf16)(s * sinf(c * (2.0f * t + 1.0f)));
  }
  *reinterpret_cast<v8bf*>(Wb + (size_t)o * IN + t0) = w;
}

// ---------------------------------------------------------------------------
// Prepass 2: x f32 -> bf16 (row-major [N][IN]), 8 elements per thread.
// ---------------------------------------------------------------------------
__global__ __launch_bounds__(256) void convert_x_bf16(
    const float* __restrict__ x, bf16* __restrict__ Xb, int totalChunks) {
  int idx = blockIdx.x * blockDim.x + threadIdx.x;
  if (idx >= totalChunks) return;
  size_t base = (size_t)idx * 8;
  v4f x0 = *reinterpret_cast<const v4f*>(x + base);
  v4f x1 = *reinterpret_cast<const v4f*>(x + base + 4);
  v8bf o;
#pragma unroll
  for (int j = 0; j < 4; ++j) { o[j] = (__bf16)x0[j]; o[j + 4] = (__bf16)x1[j]; }
  *reinterpret_cast<v8bf*>(Xb + base) = o;
}

// ---------------------------------------------------------------------------
// GEMM: y[n,o] = sum_k Xb[n,k] * Wb[o,k] + bias[o]
//
// Block = 256 threads = 8 waves, arranged 2 (M) x 4 (N).
// Wave register tile: 32 (M) x 64 (N) = 2 x 4 accumulators of
// v_wmma_f32_16x16x32_bf16; per 32-K step a wave loads 2 A frags + 4 B frags
// and issues 8 WMMAs. Fragments are software-pipelined (next K-step's loads
// issue before the current WMMAs) so matrix ops overlap the L2 round-trip.
//
// Fragment layouts per CDNA5 ISA 7.12.2 (lo = lane%16, hi = lane/16):
//   A 16x32: row M=lo; elems 0..7  = K in [hi*8,  hi*8+8)
//                      elems 8..15 = K in [16+hi*8, 16+hi*8+8)
//   B 32x16: col N=lo; elems 0..7  = K in [hi*8,  hi*8+8)
//                      elems 8..15 = K in [16+hi*8, 16+hi*8+8)
//   C 16x16: elem r -> row hi*8+r, col lo
// ---------------------------------------------------------------------------
#define MT 2   // 2 x 16 output rows per wave
#define NT 4   // 4 x 16 output cols per wave

__device__ __forceinline__ v16bf load_frag(const bf16* p0, const bf16* p1) {
  v8bf a = *reinterpret_cast<const v8bf*>(p0);
  v8bf b = *reinterpret_cast<const v8bf*>(p1);
  return __builtin_shufflevector(a, b, 0, 1, 2, 3, 4, 5, 6, 7,
                                       8, 9, 10, 11, 12, 13, 14, 15);
}

__global__ __launch_bounds__(256) void dst_gemm_wmma(
    const bf16* __restrict__ Xb, const bf16* __restrict__ Wb,
    const float* __restrict__ bias, float* __restrict__ out,
    int IN, int OUT) {
  const int lane = threadIdx.x & 31;
  const int wave = threadIdx.x >> 5;
  const int mW   = wave & 1;   // wave position in M (0..1)
  const int nW   = wave >> 1;  // wave position in N (0..3)
  const int lo   = lane & 15;
  const int hi   = lane >> 4;

  const int mBase = blockIdx.y * 64  + mW * (MT * 16);
  const int nBase = blockIdx.x * 256 + nW * (NT * 16);

  // Per-lane base pointers with the lane's K sub-offset folded in.
  const bf16* aRow[MT];
#pragma unroll
  for (int i = 0; i < MT; ++i)
    aRow[i] = Xb + (size_t)(mBase + i * 16 + lo) * IN + hi * 8;

  const bf16* bRow[NT];
  float bv[NT];
#pragma unroll
  for (int t = 0; t < NT; ++t) {
    int n = nBase + t * 16 + lo;
    bRow[t] = Wb + (size_t)n * IN + hi * 8;
    bv[t]   = bias[n];
  }

  v8f zero = {0.f, 0.f, 0.f, 0.f, 0.f, 0.f, 0.f, 0.f};
  v8f acc[MT][NT];
#pragma unroll
  for (int i = 0; i < MT; ++i)
#pragma unroll
    for (int t = 0; t < NT; ++t) acc[i][t] = zero;

  // ---- software pipeline: prologue loads for kk = 0 ----
  v16bf a[MT], b[NT];
#pragma unroll
  for (int i = 0; i < MT; ++i) a[i] = load_frag(aRow[i], aRow[i] + 16);
#pragma unroll
  for (int t = 0; t < NT; ++t) b[t] = load_frag(bRow[t], bRow[t] + 16);

#pragma unroll 2
  for (int kk = 32; kk < IN; kk += 32) {
    // issue next K-step's fragment loads first (overlap with WMMAs below)
    v16bf an[MT], bn[NT];
#pragma unroll
    for (int i = 0; i < MT; ++i) an[i] = load_frag(aRow[i] + kk, aRow[i] + kk + 16);
#pragma unroll
    for (int t = 0; t < NT; ++t) bn[t] = load_frag(bRow[t] + kk, bRow[t] + kk + 16);

    // current K-step: 8 WMMAs on already-resident fragments
#pragma unroll
    for (int i = 0; i < MT; ++i)
#pragma unroll
      for (int t = 0; t < NT; ++t)
        acc[i][t] = __builtin_amdgcn_wmma_f32_16x16x32_bf16(
            false, a[i], false, b[t], (short)0, acc[i][t], false, false);

#pragma unroll
    for (int i = 0; i < MT; ++i) a[i] = an[i];
#pragma unroll
    for (int t = 0; t < NT; ++t) b[t] = bn[t];
  }

  // ---- pipeline epilogue: last K-step ----
#pragma unroll
  for (int i = 0; i < MT; ++i)
#pragma unroll
    for (int t = 0; t < NT; ++t)
      acc[i][t] = __builtin_amdgcn_wmma_f32_16x16x32_bf16(
          false, a[i], false, b[t], (short)0, acc[i][t], false, false);

  // ---- epilogue: bias add + stores (16 lanes write 64B per row segment) ----
#pragma unroll
  for (int i = 0; i < MT; ++i) {
#pragma unroll
    for (int t = 0; t < NT; ++t) {
      int n = nBase + t * 16 + lo;
#pragma unroll
      for (int r = 0; r < 8; ++r) {
        out[(size_t)(mBase + i * 16 + hi * 8 + r) * OUT + n] = acc[i][t][r] + bv[t];
      }
    }
  }
}

// ---------------------------------------------------------------------------
// Host launcher
// ---------------------------------------------------------------------------
extern "C" void kernel_launch(void* const* d_in, const int* in_sizes, int n_in,
                              void* d_out, int out_size, void* d_ws, size_t ws_size,
                              hipStream_t stream) {
  const float* x    = (const float*)d_in[0];  // [N, IN] f32
  const float* fc   = (const float*)d_in[1];  // [OUT, 1] f32
  const float* bias = (const float*)d_in[2];  // [OUT] f32
  float* out = (float*)d_out;                 // [N, OUT] f32

  const int OUT = in_sizes[2];                // 4096
  const int N   = out_size / OUT;             // 8192
  const int IN  = in_sizes[0] / N;            // 4096

  // Workspace layout: [ W_bf16 : OUT*IN ][ X_bf16 : N*IN ]  (~96 MiB total)
  bf16* Wb = (bf16*)d_ws;
  bf16* Xb = (bf16*)((char*)d_ws + (size_t)OUT * IN * sizeof(bf16));

  {
    int total = OUT * (IN >> 3);
    build_w_bf16<<<(total + 255) / 256, 256, 0, stream>>>(fc, Wb, OUT, IN);
  }
  {
    int total = (int)(((size_t)N * IN) >> 3);
    convert_x_bf16<<<(total + 255) / 256, 256, 0, stream>>>(x, Xb, total);
  }
  {
    dim3 grid(OUT / 256, N / 64);  // 16 x 128 workgroups
    dst_gemm_wmma<<<grid, 256, 0, stream>>>(Xb, Wb, bias, out, IN, OUT);
  }
}
